// SimpleKANLayer_60945585930797
// MI455X (gfx1250) — matrix operand order repaired and checked by reading, not compiled, not run on hardware
//
#include <hip/hip_runtime.h>

typedef __attribute__((ext_vector_type(16))) _Float16 v16h;
typedef __attribute__((ext_vector_type(8)))  _Float16 v8h;
typedef __attribute__((ext_vector_type(4)))  _Float16 v4h;
typedef __attribute__((ext_vector_type(8)))  float    v8f;
typedef __attribute__((ext_vector_type(4)))  float    v4f;

typedef __attribute__((address_space(3))) _Float16 lds_h;

#define BATCH 8192
#define IDIM  768
#define JDIM  768
#define GRIDN 5
#define KDIM  (GRIDN * IDIM)  // 3840 ; k = g*IDIM + i  (g-major)
#define TM    128
#define TN    128
#define KT    128             // K tile (4 wmma k-steps of 32); 768 % 128 == 0 so a tile never crosses g
#define LDA   136             // padded LDS stride in halfs (272 B -> bank-conflict-free b128 reads)

// ---------------- prep 1: xt = f16(tanh(x)) ----------------
__global__ void kan_tanh_kernel(const float* __restrict__ x, _Float16* __restrict__ xt) {
    size_t idx = (size_t)blockIdx.x * blockDim.x + threadIdx.x;   // one float4 per thread
    v4f xv = ((const v4f*)x)[idx];
    v4h tv;
#pragma unroll
    for (int e = 0; e < 4; ++e) {
        // tanh(x) = 1 - 2/(exp(2x)+1); exp(2x) = 2^(2*log2e*x). Safe at +-inf.
        float ex = __builtin_exp2f(2.885390081777927f * xv[e]);
        float t  = 1.0f - 2.0f * __builtin_amdgcn_rcpf(ex + 1.0f);
        tv[e] = (_Float16)t;
    }
    ((v4h*)xt)[idx] = tv;
}

// ---------------- prep 2: Wt[j][g*768+i] = f16(coeffs[i][j][g] * scale[i][j]) ----------------
__global__ void kan_wprep_kernel(const float* __restrict__ coeffs, const float* __restrict__ scale,
                                 _Float16* __restrict__ wt) {
    int tid = blockIdx.x * 256 + threadIdx.x;      // 768*768 threads
    int i = tid % IDIM;
    int j = tid / IDIM;
    float s = scale[(size_t)i * JDIM + j];
    const float* cp = coeffs + ((size_t)i * JDIM + j) * GRIDN;
#pragma unroll
    for (int g = 0; g < GRIDN; ++g) {
        wt[(size_t)j * KDIM + g * IDIM + i] = (_Float16)(cp[g] * s);
    }
}

// ---------------- prep 3: bias[j] = sum_i shift[i][j] ----------------
__global__ void kan_bias_kernel(const float* __restrict__ shift, float* __restrict__ bias) {
    int j = blockIdx.x * 256 + threadIdx.x;
    if (j < JDIM) {
        float s = 0.0f;
        for (int i = 0; i < IDIM; ++i) s += shift[(size_t)i * JDIM + j];
        bias[j] = s;
    }
}

static __device__ __forceinline__ v16h frag_combine(v8h lo, v8h hi) {
    v16h r;
#pragma unroll
    for (int e = 0; e < 8; ++e) { r[e] = lo[e]; r[e + 8] = hi[e]; }
    return r;
}

// gfx1250 async global->LDS copy, 16 bytes per lane (ASYNCcnt-tracked, no VGPR round trip)
static __device__ __forceinline__ void async_copy_b128(_Float16* lds_dst, const _Float16* gsrc) {
    lds_h* l = (lds_h*)lds_dst;
    asm volatile("global_load_async_to_lds_b128 %0, %1, off"
                 :: "v"(l), "v"(gsrc) : "memory");
}
static __device__ __forceinline__ void wait_async0() {
    asm volatile("s_wait_asynccnt 0x0" ::: "memory");
}

// ---------------- main GEMM: out[b][j] = sum_k basis[b][k] * Wt[j][k] + bias[j] ----------------
__global__ __launch_bounds__(256)
void kan_gemm_kernel(const _Float16* __restrict__ xt, const _Float16* __restrict__ wt,
                     const float* __restrict__ bias, float* __restrict__ out) {
    __shared__ _Float16 As[2][TM * LDA];   // basis tiles   [row][k]  2 x 34816 B
    __shared__ _Float16 Bs[2][TN * LDA];   // weight tiles  [col][k]  2 x 34816 B

    const int b0   = blockIdx.x * TM;
    const int j0   = blockIdx.y * TN;
    const int tid  = threadIdx.x;
    const int wave = tid >> 5;
    const int lane = tid & 31;
    const int wm   = wave >> 1;          // 0..3 : 32-row slab
    const int wn   = wave & 1;           // 0..1 : 64-col slab
    const int lrow = lane & 15;
    const int kba  = (lane < 16) ? 0 : 8;    // A fragment K base (halfs)
    const int kbb  = (lane < 16) ? 0 : 8;    // B fragment K base (halfs)

    // fill-stage mapping: 16 chunks of 8 halfs along K/I, 16 rows per pass
    const int chunk = tid & 15;
    const int rbase = tid >> 4;

    v8f acc[2][4] = {};

    // -------- tile fill helpers --------
    auto fillA = [&](int buf, int i0, float gg) {
#pragma unroll
        for (int p = 0; p < 8; ++p) {
            const int r = rbase + p * 16;
            v8h xv = *(const v8h*)(xt + (size_t)(b0 + r) * IDIM + i0 + chunk * 8);
            v8h bv;
#pragma unroll
            for (int e = 0; e < 8; ++e) {
                float d = (float)xv[e] - gg;
                // exp(-5 d^2) = 2^(-5*log2e * d^2)
                bv[e] = (_Float16)__builtin_exp2f(-7.213475204444817f * d * d);
            }
            *(v8h*)(&As[buf][r * LDA + chunk * 8]) = bv;
        }
    };
    auto fillB = [&](int buf, int kt) {
#pragma unroll
        for (int p = 0; p < 8; ++p) {
            const int n = rbase + p * 16;
            async_copy_b128(&Bs[buf][n * LDA + chunk * 8],
                            wt + (size_t)(j0 + n) * KDIM + kt + chunk * 8);
        }
    };
    auto compute = [&](int buf) {
#pragma unroll
        for (int ks = 0; ks < KT / 32; ++ks) {
            v16h afrag[2];
            v16h bfrag[4];
#pragma unroll
            for (int ms = 0; ms < 2; ++ms) {
                const _Float16* ap = &As[buf][(wm * 32 + ms * 16 + lrow) * LDA + ks * 32 + kba];
                afrag[ms] = frag_combine(*(const v8h*)ap, *(const v8h*)(ap + 16));
            }
#pragma unroll
            for (int ns = 0; ns < 4; ++ns) {
                const _Float16* bp = &Bs[buf][(wn * 64 + ns * 16 + lrow) * LDA + ks * 32 + kbb];
                bfrag[ns] = frag_combine(*(const v8h*)bp, *(const v8h*)(bp + 16));
            }
#pragma unroll
            for (int ms = 0; ms < 2; ++ms)
#pragma unroll
                for (int ns = 0; ns < 4; ++ns)
                    acc[ms][ns] = __builtin_amdgcn_wmma_f32_16x16x32_f16(
                        false, afrag[ms], false, bfrag[ns],
                        (short)0, acc[ms][ns], false, false);
        }
    };

    // -------- double-buffered main loop --------
    const int NT = KDIM / KT;   // 30
    fillA(0, 0, -1.0f);
    fillB(0, 0);
    wait_async0();
    __syncthreads();

    for (int t = 0; t < NT; ++t) {
        if (t + 1 < NT) {
            const int kt1 = (t + 1) * KT;
            const int g1  = kt1 / IDIM;
            const int i1  = kt1 - g1 * IDIM;
            fillA((t + 1) & 1, i1, -1.0f + 0.5f * (float)g1);   // VALU/TRANS overlaps WMMA
            fillB((t + 1) & 1, kt1);                            // async DMA overlaps WMMA
        }
        compute(t & 1);
        wait_async0();       // drain async copies into the other buffer
        __syncthreads();     // publish LDS stores (compiler adds dscnt wait) + sync waves
    }

    // ---- epilogue: D layout => lane l: N = l%16 ; M = v + (l<16 ? 0 : 8) ----
    float bv[4];
#pragma unroll
    for (int ns = 0; ns < 4; ++ns) bv[ns] = bias[j0 + wn * 64 + ns * 16 + lrow];
    const int mofs = (lane < 16) ? 0 : 8;
#pragma unroll
    for (int ms = 0; ms < 2; ++ms)
#pragma unroll
        for (int ns = 0; ns < 4; ++ns) {
            const int col = j0 + wn * 64 + ns * 16 + lrow;
#pragma unroll
            for (int v = 0; v < 8; ++v) {
                const int row = b0 + wm * 32 + ms * 16 + mofs + v;
                out[(size_t)row * JDIM + col] = acc[ms][ns][v] + bv[ns];
            }
        }
}

extern "C" void kernel_launch(void* const* d_in, const int* in_sizes, int n_in,
                              void* d_out, int out_size, void* d_ws, size_t ws_size,
                              hipStream_t stream) {
    (void)in_sizes; (void)n_in; (void)out_size; (void)ws_size;
    const float* x      = (const float*)d_in[0];
    const float* coeffs = (const float*)d_in[1];
    const float* scale  = (const float*)d_in[2];
    const float* shift  = (const float*)d_in[3];
    float* out = (float*)d_out;

    char* ws = (char*)d_ws;
    _Float16* wt   = (_Float16*)ws;                                    // 768*3840*2 = 5,898,240 B
    _Float16* xt   = (_Float16*)(ws + 5898240);                        // 8192*768*2 = 12,582,912 B
    float*    bias = (float*)(ws + 5898240 + 12582912);                // 3,072 B

    kan_tanh_kernel <<<(BATCH * IDIM / 4) / 256, 256, 0, stream>>>(x, xt);
    kan_wprep_kernel<<<(IDIM * JDIM) / 256,      256, 0, stream>>>(coeffs, scale, wt);
    kan_bias_kernel <<<3,                        256, 0, stream>>>(shift, bias);

    dim3 grid(BATCH / TM, JDIM / TN);   // 64 x 6
    kan_gemm_kernel<<<grid, 256, 0, stream>>>(xt, wt, bias, out);
}
